// CausalGatedLinearAttentionV10_70583492542996
// MI455X (gfx1250) — matrix operand order repaired and checked by reading, not compiled, not run on hardware
//
#include <hip/hip_runtime.h>

#define D_MODEL 1024
#define T_SEQ   2048
#define N_HEADS 16
#define D_HEAD  64
#define CHUNK   64
#define N_CHUNKS (T_SEQ / CHUNK)   // 32

typedef __attribute__((ext_vector_type(16))) __bf16 v16bf;
typedef __attribute__((ext_vector_type(8)))  float  v8f;
typedef unsigned short u16;
typedef unsigned int   u32;

struct __align__(16) B128 { u32 x, y, z, w; };
union Frag { B128 u[2]; v16bf v; };

__device__ __forceinline__ u16 f2bf(float f) {
  u32 u = __float_as_uint(f);
  u32 r = u + 0x7FFFu + ((u >> 16) & 1u);
  return (u16)(r >> 16);
}
__device__ __forceinline__ float bf2f(u16 b) {
  return __uint_as_float(((u32)b) << 16);
}
__device__ __forceinline__ v8f wmma_bf(v16bf a, v16bf b, v8f c) {
  return __builtin_amdgcn_wmma_f32_16x16x32_bf16(false, a, false, b, (short)0, c,
                                                 false, false);
}
__device__ __forceinline__ v8f zero8() {
  v8f z = {0.f,0.f,0.f,0.f,0.f,0.f,0.f,0.f};
  return z;
}

// ---------------------------------------------------------------- LayerNorm
__global__ void ln_kernel(const float* __restrict__ x,
                          const float* __restrict__ g,
                          const float* __restrict__ b,
                          u16* __restrict__ xn_bf, u16* __restrict__ x_bf) {
  const int t = blockIdx.x, tid = threadIdx.x;
  __shared__ float red[256];
  const float* row = x + (size_t)t * D_MODEL;

  float s = 0.f;
  for (int i = tid; i < D_MODEL; i += 256) s += row[i];
  red[tid] = s; __syncthreads();
  for (int o = 128; o > 0; o >>= 1) { if (tid < o) red[tid] += red[tid + o]; __syncthreads(); }
  const float mu = red[0] * (1.f / D_MODEL);
  __syncthreads();

  float v = 0.f;
  for (int i = tid; i < D_MODEL; i += 256) { float d = row[i] - mu; v += d * d; }
  red[tid] = v; __syncthreads();
  for (int o = 128; o > 0; o >>= 1) { if (tid < o) red[tid] += red[tid + o]; __syncthreads(); }
  const float inv = rsqrtf(red[0] * (1.f / D_MODEL) + 1e-5f);

  for (int i = tid; i < D_MODEL; i += 256) {
    const float xv = row[i];
    xn_bf[(size_t)t * D_MODEL + i] = f2bf((xv - mu) * inv * g[i] + b[i]);
    x_bf [(size_t)t * D_MODEL + i] = f2bf(xv);
  }
}

// ------------------------------------------------------- f32 -> bf16 convert
__global__ void cvt_kernel(const float* __restrict__ in, u16* __restrict__ out, int n) {
  const int i = blockIdx.x * 256 + threadIdx.x;
  if (i < n) out[i] = f2bf(in[i]);
}

// ------------------------------------------------ WMMA GEMM: out = A @ W^T + b
// A: (M,K) bf16 row-major. W: (N,K) bf16 row-major. out: (M,N) f32.
// block = 128 threads = 4 waves in 2x2; block tile 128x128, wave tile 64x64.
// launch_bounds(128,1): allow full VGPR budget so 16 v8f accums stay resident.
__global__ void __launch_bounds__(128, 1)
gemm_bf16_kernel(const u16* __restrict__ A, const u16* __restrict__ W,
                 const float* __restrict__ bias, float* __restrict__ out,
                 int M, int N, int K) {
  const int tid = threadIdx.x, lane = tid & 31, wid = tid >> 5;
  const int wm = wid >> 1, wn = wid & 1;
  const int m0 = blockIdx.y * 128 + wm * 64;
  const int n0 = blockIdx.x * 128 + wn * 64;
  const int l15 = lane & 15;
  const int aoff = (lane & 16) ? 16 : 0;
  const int boff = (lane & 16) ? 16 : 0;

  v8f acc[4][4];
#pragma unroll
  for (int mi = 0; mi < 4; ++mi)
#pragma unroll
    for (int ni = 0; ni < 4; ++ni) acc[mi][ni] = zero8();

  const u16* pa0 = A + (size_t)(m0 + l15) * K + aoff;
  const u16* pb0 = W + (size_t)(n0 + l15) * K + boff;

#pragma unroll 1
  for (int k0 = 0; k0 < K; k0 += 32) {
    Frag fa[4];
#pragma unroll
    for (int mi = 0; mi < 4; ++mi) {
      const u16* p = pa0 + (size_t)(mi * 16) * K + k0;
      fa[mi].u[0] = *(const B128*)p;
      fa[mi].u[1] = *(const B128*)(p + 8);
    }
#pragma unroll
    for (int ni = 0; ni < 4; ++ni) {
      const u16* p = pb0 + (size_t)(ni * 16) * K + k0;
      Frag fb;
      fb.u[0] = *(const B128*)p;
      fb.u[1] = *(const B128*)(p + 8);
#pragma unroll
      for (int mi = 0; mi < 4; ++mi)
        acc[mi][ni] = wmma_bf(fa[mi].v, fb.v, acc[mi][ni]);
    }
  }

#pragma unroll
  for (int mi = 0; mi < 4; ++mi)
#pragma unroll
    for (int ni = 0; ni < 4; ++ni) {
      const int col = n0 + ni * 16 + l15;
      const float bv = bias ? bias[col] : 0.f;
      const int rb = m0 + mi * 16 + ((lane & 16) ? 8 : 0);
#pragma unroll
      for (int j = 0; j < 8; ++j)
        out[(size_t)(rb + j) * N + col] = acc[mi][ni][j] + bv;
    }
}

// -------------------------------------- gate: sigmoid, mean-norm, elu+1 on q,k
__global__ void gate_kernel(const float* __restrict__ qkv, const float* __restrict__ gl,
                            u16* __restrict__ qb, u16* __restrict__ kb,
                            u16* __restrict__ kT, u16* __restrict__ vT) {
  const int t = blockIdx.x, tid = threadIdx.x;
  __shared__ float red[256];
  __shared__ float gn[D_MODEL];

  float s = 0.f;
  for (int i = tid; i < D_MODEL; i += 256) {
    const float sg = 1.f / (1.f + __expf(-gl[(size_t)t * D_MODEL + i]));
    gn[i] = sg; s += sg;
  }
  red[tid] = s; __syncthreads();
  for (int o = 128; o > 0; o >>= 1) { if (tid < o) red[tid] += red[tid + o]; __syncthreads(); }
  const float scale = 1.f / (red[0] * (1.f / D_MODEL) + 1e-5f);

  const float* qrow = qkv + (size_t)t * 3 * D_MODEL;
  for (int i = tid; i < D_MODEL; i += 256) {
    const float gv = gn[i] * scale;
    float q = qrow[i] * gv;
    float k = qrow[D_MODEL + i] * gv;
    const float v = qrow[2 * D_MODEL + i];
    q = (q > 0.f) ? q + 1.f : __expf(q);   // elu(q)+1
    k = (k > 0.f) ? k + 1.f : __expf(k);
    qb[(size_t)t * D_MODEL + i] = f2bf(q);
    kb[(size_t)t * D_MODEL + i] = f2bf(k);
    kT[(size_t)i * T_SEQ + t] = f2bf(k);
    vT[(size_t)i * T_SEQ + t] = f2bf(v);
  }
}

// ------------------------- per-(head,chunk) state: ST = V^T @ K, ksum(K chunk)
__global__ void __launch_bounds__(128, 1)
chunk_st_kernel(const u16* __restrict__ kT, const u16* __restrict__ vT,
                const u16* __restrict__ kb,
                float* __restrict__ ST, float* __restrict__ ksum) {
  const int c = blockIdx.x, h = blockIdx.y;
  const int tid = threadIdx.x, lane = tid & 31, wid = tid >> 5;
  const int t0 = c * CHUNK;
  const int l15 = lane & 15;
  const int aoff = (lane & 16) ? 16 : 0;
  const int boff = (lane & 16) ? 16 : 0;

  v8f acc[4];
#pragma unroll
  for (int ni = 0; ni < 4; ++ni) acc[ni] = zero8();

  const int arow = h * D_HEAD + wid * 16 + l15;     // v-dim row of V^T
#pragma unroll
  for (int ks = 0; ks < 2; ++ks) {
    Frag fa;
    const u16* pa = vT + (size_t)arow * T_SEQ + t0 + ks * 32 + aoff;
    fa.u[0] = *(const B128*)pa;
    fa.u[1] = *(const B128*)(pa + 8);
#pragma unroll
    for (int ni = 0; ni < 4; ++ni) {
      const int nrow = h * D_HEAD + ni * 16 + l15;  // k-dim row of K^T
      const u16* pb = kT + (size_t)nrow * T_SEQ + t0 + ks * 32 + boff;
      Frag fb;
      fb.u[0] = *(const B128*)pb;
      fb.u[1] = *(const B128*)(pb + 8);
      acc[ni] = wmma_bf(fa.v, fb.v, acc[ni]);
    }
  }

  float* stb = ST + ((size_t)(h * N_CHUNKS + c)) * (D_HEAD * D_HEAD);
#pragma unroll
  for (int ni = 0; ni < 4; ++ni) {
    const int col = ni * 16 + l15;
    const int rb = wid * 16 + ((lane & 16) ? 8 : 0);
#pragma unroll
    for (int j = 0; j < 8; ++j)
      stb[(size_t)(rb + j) * D_HEAD + col] = acc[ni][j];
  }

  if (tid < D_HEAD) {
    float s = 0.f;
    for (int t = 0; t < CHUNK; ++t)
      s += bf2f(kb[(size_t)(t0 + t) * D_MODEL + h * D_HEAD + tid]);
    ksum[(size_t)(h * N_CHUNKS + c) * D_HEAD + tid] = s;
  }
}

// -------------------- exclusive prefix over chunks: STp (bf16), kpre (f32)
__global__ void prefix_kernel(const float* __restrict__ ST, const float* __restrict__ ksum,
                              u16* __restrict__ STp, float* __restrict__ kpre) {
  const int h = blockIdx.x, tid = threadIdx.x;
  float acc[16];
#pragma unroll
  for (int e = 0; e < 16; ++e) acc[e] = 0.f;
  for (int c = 0; c < N_CHUNKS; ++c) {
    const size_t base = ((size_t)(h * N_CHUNKS + c)) * (D_HEAD * D_HEAD);
#pragma unroll
    for (int e = 0; e < 16; ++e) {
      const int pos = tid + e * 256;
      STp[base + pos] = f2bf(acc[e]);
      acc[e] += ST[base + pos];
    }
  }
  if (tid < D_HEAD) {
    float a = 0.f;
    for (int c = 0; c < N_CHUNKS; ++c) {
      const size_t base = (size_t)(h * N_CHUNKS + c) * D_HEAD + tid;
      kpre[base] = a;
      a += ksum[base];
    }
  }
}

// ------------- per-(head,chunk): out = mask(QK^T)V + Q Sprev, divide by denom
__global__ void __launch_bounds__(128, 1)
attn_kernel(const u16* __restrict__ qb, const u16* __restrict__ kb,
            const u16* __restrict__ vT, const u16* __restrict__ STp,
            const float* __restrict__ kpre, u16* __restrict__ attn) {
  const int c = blockIdx.x, h = blockIdx.y;
  const int tid = threadIdx.x, lane = tid & 31, wid = tid >> 5;
  const int t0 = c * CHUNK;
  const int l15 = lane & 15;
  const int aoff = (lane & 16) ? 16 : 0;
  const int boff = (lane & 16) ? 16 : 0;

  __shared__ __align__(16) u16 P[CHUNK * CHUNK];
  __shared__ float rs[CHUNK];
  __shared__ float den[CHUNK];
  __shared__ float kl[CHUNK];

  // Q A-fragments (kept for both score and Q@S passes)
  Frag aq[2];
  {
    const int qrow = t0 + wid * 16 + l15;
    const u16* pq = qb + (size_t)qrow * D_MODEL + h * D_HEAD + aoff;
    aq[0].u[0] = *(const B128*)pq;
    aq[0].u[1] = *(const B128*)(pq + 8);
    aq[1].u[0] = *(const B128*)(pq + 32);
    aq[1].u[1] = *(const B128*)(pq + 40);
  }

  // scores = Q @ K^T  (intra-chunk), 16x64 per wave
  v8f sc[4];
#pragma unroll
  for (int ni = 0; ni < 4; ++ni) sc[ni] = zero8();
#pragma unroll
  for (int ks = 0; ks < 2; ++ks)
#pragma unroll
    for (int ni = 0; ni < 4; ++ni) {
      const int jrow = t0 + ni * 16 + l15;
      const u16* pk = kb + (size_t)jrow * D_MODEL + h * D_HEAD + ks * 32 + boff;
      Frag fb;
      fb.u[0] = *(const B128*)pk;
      fb.u[1] = *(const B128*)(pk + 8);
      sc[ni] = wmma_bf(aq[ks].v, fb.v, sc[ni]);
    }

  // causal mask (j <= i), row-sums, stage masked P (bf16) to LDS
  const int jcol = l15;
  const int ibase = wid * 16 + ((lane & 16) ? 8 : 0);
  float rsum[8];
#pragma unroll
  for (int j = 0; j < 8; ++j) rsum[j] = 0.f;
#pragma unroll
  for (int ni = 0; ni < 4; ++ni) {
    const int j = ni * 16 + jcol;
#pragma unroll
    for (int vj = 0; vj < 8; ++vj) {
      const int i = ibase + vj;
      float v = sc[ni][vj];
      if (j > i) v = 0.f;
      rsum[vj] += v;
      P[i * CHUNK + j] = f2bf(v);
    }
  }
#pragma unroll
  for (int vj = 0; vj < 8; ++vj) {
    float v = rsum[vj];
    for (int off = 1; off < 16; off <<= 1) v += __shfl_xor(v, off, 32);
    if ((lane & 15) == 0) rs[ibase + vj] = v;
  }
  if (tid < D_HEAD) kl[tid] = kpre[(size_t)(h * N_CHUNKS + c) * D_HEAD + tid];
  __syncthreads();

  // denominator: rowsum + q . ksum_prefix + eps
  if (tid < CHUNK) {
    const u16* pq2 = qb + (size_t)(t0 + tid) * D_MODEL + h * D_HEAD;
    float dot = 0.f;
    for (int d = 0; d < D_HEAD; ++d) dot += bf2f(pq2[d]) * kl[d];
    den[tid] = dot + rs[tid] + 1e-5f;
  }

  // out = P @ V + Q @ S_prefix, 16x64 per wave
  v8f oacc[4];
#pragma unroll
  for (int ni = 0; ni < 4; ++ni) oacc[ni] = zero8();
  const u16* stb = STp + ((size_t)(h * N_CHUNKS + c)) * (D_HEAD * D_HEAD);
#pragma unroll
  for (int ks = 0; ks < 2; ++ks) {
    Frag ap;
    const u16* pp = &P[(wid * 16 + l15) * CHUNK + ks * 32 + aoff];
    ap.u[0] = *(const B128*)pp;
    ap.u[1] = *(const B128*)(pp + 8);
#pragma unroll
    for (int ni = 0; ni < 4; ++ni) {
      const int nrow = h * D_HEAD + ni * 16 + l15;
      const u16* pv = vT + (size_t)nrow * T_SEQ + t0 + ks * 32 + boff;
      Frag fv;
      fv.u[0] = *(const B128*)pv;
      fv.u[1] = *(const B128*)(pv + 8);
      oacc[ni] = wmma_bf(ap.v, fv.v, oacc[ni]);

      const u16* ps = stb + (size_t)(ni * 16 + l15) * D_HEAD + ks * 32 + boff;
      Frag fs;
      fs.u[0] = *(const B128*)ps;
      fs.u[1] = *(const B128*)(ps + 8);
      oacc[ni] = wmma_bf(aq[ks].v, fs.v, oacc[ni]);
    }
  }
  __syncthreads();

#pragma unroll
  for (int ni = 0; ni < 4; ++ni) {
    const int m = ni * 16 + jcol;
#pragma unroll
    for (int j = 0; j < 8; ++j) {
      const int i = ibase + j;
      attn[(size_t)(t0 + i) * D_MODEL + h * D_HEAD + m] = f2bf(oacc[ni][j] / den[i]);
    }
  }
}

// ------------------------------------------------------------------- launch
extern "C" void kernel_launch(void* const* d_in, const int* in_sizes, int n_in,
                              void* d_out, int out_size, void* d_ws, size_t ws_size,
                              hipStream_t stream) {
  const float* x      = (const float*)d_in[0];
  const float* ln_g   = (const float*)d_in[1];
  const float* ln_b   = (const float*)d_in[2];
  const float* qkv_w  = (const float*)d_in[3];
  const float* qkv_b  = (const float*)d_in[4];
  const float* gate_w = (const float*)d_in[5];
  const float* gate_b = (const float*)d_in[6];
  const float* proj_w = (const float*)d_in[7];
  const float* proj_b = (const float*)d_in[8];
  float* out = (float*)d_out;

  char* w = (char*)d_ws;
  auto take = [&](size_t bytes) -> void* {
    void* p = (void*)w;
    w += (bytes + 255) & ~(size_t)255;
    return p;
  };
  const size_t TD = (size_t)T_SEQ * D_MODEL;
  u16*   xbf    = (u16*)take(TD * 2);
  u16*   xnbf   = (u16*)take(TD * 2);
  u16*   qkvwb  = (u16*)take((size_t)3 * D_MODEL * D_MODEL * 2);
  u16*   gatewb = (u16*)take((size_t)D_MODEL * D_MODEL * 2);
  u16*   projwb = (u16*)take((size_t)D_MODEL * D_MODEL * 2);
  float* qkvf   = (float*)take(TD * 3 * 4);
  float* gatef  = (float*)take(TD * 4);
  u16*   qbf    = (u16*)take(TD * 2);
  u16*   kbf    = (u16*)take(TD * 2);
  u16*   kTb    = (u16*)take(TD * 2);
  u16*   vTb    = (u16*)take(TD * 2);
  float* ST     = (float*)take((size_t)N_HEADS * N_CHUNKS * D_HEAD * D_HEAD * 4);
  u16*   STp    = (u16*)take((size_t)N_HEADS * N_CHUNKS * D_HEAD * D_HEAD * 2);
  float* ksum   = (float*)take((size_t)N_HEADS * N_CHUNKS * D_HEAD * 4);
  float* kpre   = (float*)take((size_t)N_HEADS * N_CHUNKS * D_HEAD * 4);
  u16*   attnb  = (u16*)take(TD * 2);

  // 1. LayerNorm + bf16 casts of x / x_norm
  ln_kernel<<<T_SEQ, 256, 0, stream>>>(x, ln_g, ln_b, xnbf, xbf);

  // 2. weight conversions
  {
    int n = 3 * D_MODEL * D_MODEL;
    cvt_kernel<<<(n + 255) / 256, 256, 0, stream>>>(qkv_w, qkvwb, n);
    n = D_MODEL * D_MODEL;
    cvt_kernel<<<(n + 255) / 256, 256, 0, stream>>>(gate_w, gatewb, n);
    cvt_kernel<<<(n + 255) / 256, 256, 0, stream>>>(proj_w, projwb, n);
  }

  // 3. QKV + gate projections (WMMA)
  gemm_bf16_kernel<<<dim3(3 * D_MODEL / 128, T_SEQ / 128), 128, 0, stream>>>(
      xnbf, qkvwb, qkv_b, qkvf, T_SEQ, 3 * D_MODEL, D_MODEL);
  gemm_bf16_kernel<<<dim3(D_MODEL / 128, T_SEQ / 128), 128, 0, stream>>>(
      xbf, gatewb, gate_b, gatef, T_SEQ, D_MODEL, D_MODEL);

  // 4. gate activation + elu+1, emit q,k (row-major) and k^T, v^T
  gate_kernel<<<T_SEQ, 256, 0, stream>>>(qkvf, gatef, qbf, kbf, kTb, vTb);

  // 5. per-chunk state ST = V^T K and k column sums (WMMA)
  chunk_st_kernel<<<dim3(N_CHUNKS, N_HEADS), 128, 0, stream>>>(kTb, vTb, kbf, ST, ksum);

  // 6. exclusive prefix over chunks
  prefix_kernel<<<N_HEADS, 256, 0, stream>>>(ST, ksum, STp, kpre);

  // 7. attention output per chunk (WMMA)
  attn_kernel<<<dim3(N_CHUNKS, N_HEADS), 128, 0, stream>>>(qbf, kbf, vTb, STp, kpre, attnb);

  // 8. output projection (WMMA)
  gemm_bf16_kernel<<<dim3(D_MODEL / 128, T_SEQ / 128), 128, 0, stream>>>(
      attnb, projwb, proj_b, out, T_SEQ, D_MODEL, D_MODEL);
}
